// LlamaAttention_65721589563914
// MI455X (gfx1250) — matrix operand order repaired and checked by reading, not compiled, not run on hardware
//
#include <hip/hip_runtime.h>

// ---------------------------------------------------------------------------
// MI455X (gfx1250) LlamaAttention decode step, bf16 WMMA throughout.
// B=8 Q=16 P=4096 H=4096 NH=32 NKV=8 D=128, S=4112.
// HBM-bound on the 268MB fp32 KV cache (read once): flash-decoding with a
// 4-way S-split, Tensor Data Mover double-buffering of K/V tiles into LDS,
// all GEMMs via v_wmma_f32_16x16x32_bf16 (f32 accumulate).
// ---------------------------------------------------------------------------

typedef __attribute__((ext_vector_type(16))) __bf16 v16bf;
typedef __attribute__((ext_vector_type(8)))  __bf16 v8bf;
typedef __attribute__((ext_vector_type(8)))  float  v8f;
typedef __attribute__((ext_vector_type(4)))  unsigned int v4u;
typedef __attribute__((ext_vector_type(8)))  int v8i;
typedef __attribute__((ext_vector_type(4)))  int v4i;

union V16U { v16bf v; v8bf h[2]; };

#define B_    8
#define Q_    16
#define P_    4096
#define H_    4096
#define NH_   32
#define NKV_  8
#define D_    128
#define S_    (P_ + Q_)      // 4112
#define ROWS  (B_ * Q_)      // 128
#define GQA   4              // NH/NKV
#define NSPLIT 4
#define KV_TILE 64
#define NTILES ((S_ + KV_TILE - 1) / KV_TILE)   // 65
#define TPS 17               // ceil(65/4) tiles per split

__device__ __forceinline__ v8f wmma_bf16(v16bf a, v16bf b, v8f c) {
  // 8 args: (neg_a, A, neg_b, B, c_mod, C, reuse_a, reuse_b)
  return __builtin_amdgcn_wmma_f32_16x16x32_bf16(false, a, false, b, (short)0, c,
                                                 false, false);
}

// ---------------------------------------------------------------------------
// TDM: issue a 2D tensor_load_to_lds of a KV_TILE x 128 fp32 tile.
// D# packed per CDNA5 ISA ch.8: group0 = {count=1, lds_addr, global_addr,
// type=2}; group1 = {mask=0, data_size=4B, tensor_dim0=128, tensor_dim1=64,
// tile_dim0=128, tile_dim1=64, tensor_dim0_stride=128}. Trailing groups zero
// (2D tensor). This toolchain's builtin takes 6 args (extra int32x8 group).
// ---------------------------------------------------------------------------
__device__ __forceinline__ void tdm_load_tile_f32(unsigned lds_byte_off,
                                                  const float* gbase) {
  const unsigned td0 = 128, td1 = KV_TILE, tile0 = 128, tile1 = KV_TILE;
  const unsigned long long ga = (unsigned long long)(size_t)gbase;
  const unsigned long long s0 = 128;  // row stride in elements
  v4u g0;
  g0[0] = 1u;                                            // count=1 (valid D#)
  g0[1] = lds_byte_off;                                  // lds_addr [63:32]
  g0[2] = (unsigned)ga;                                  // global_addr lo
  g0[3] = (unsigned)((ga >> 32) & 0x01FFFFFFull) | (2u << 30);  // hi | type=2
  v8i g1;
  g1[0] = (int)(2u << 16);                               // data_size=2 -> 4B
  g1[1] = (int)((td0 & 0xFFFFu) << 16);                  // tensor_dim0[15:0]
  g1[2] = (int)(((td0 >> 16) & 0xFFFFu) | ((td1 & 0xFFFFu) << 16));
  g1[3] = (int)(((td1 >> 16) & 0xFFFFu) | ((tile0 & 0xFFFFu) << 16));
  g1[4] = (int)(tile1 & 0xFFFFu);                        // tile_dim1, tile_dim2=0
  g1[5] = (int)(unsigned)s0;                             // dim0_stride[31:0]
  g1[6] = (int)((s0 >> 32) & 0xFFFFu);                   // dim0_stride[47:32]
  g1[7] = 0;
  v4i gz4 = {0, 0, 0, 0};
  v8i gz8 = {0, 0, 0, 0, 0, 0, 0, 0};
  __builtin_amdgcn_tensor_load_to_lds(g0, g1, gz4, gz4, gz8, 0);
}

// ---------------------------------------------------------------------------
// GEMM: C[128, N] = A[128, K] * Bw[K, N], A fp32-or-bf16, Bw fp32 (converted
// to bf16 inline; each column read exactly once across the grid, A stays in
// L2). 4 waves/block; block owns 32 columns; wave owns 32 rows x 32 cols.
// ---------------------------------------------------------------------------
template <typename AT>
__global__ __launch_bounds__(128)
void gemm128_wmma(const AT* __restrict__ A, const float* __restrict__ Bw,
                  float* __restrict__ C, int K, int N) {
  __shared__ __attribute__((aligned(16))) __bf16 Alds[128 * 32];
  __shared__ __attribute__((aligned(16))) __bf16 BTlds[32 * 32];
  const int tid  = threadIdx.x;
  const int wave = tid >> 5, lane = tid & 31;
  const int l16  = lane & 15, half = lane >> 4;
  const int col0 = blockIdx.x * 32;

  v8f acc[2][2] = {};

  for (int k0 = 0; k0 < K; k0 += 32) {
    __syncthreads();
    for (int i = tid; i < 128 * 32; i += 128) {
      int r = i >> 5, d = i & 31;
      Alds[i] = (__bf16)(float)A[(size_t)r * K + k0 + d];
    }
    for (int i = tid; i < 32 * 32; i += 128) {
      int n = i & 31, kk = i >> 5;
      BTlds[n * 32 + kk] = (__bf16)Bw[(size_t)(k0 + kk) * N + col0 + n];
    }
    __syncthreads();

    V16U bfrag[2];
#pragma unroll
    for (int nt = 0; nt < 2; ++nt) {
      const __bf16* bp = BTlds + (nt * 16 + l16) * 32 + 16 * half;
      bfrag[nt].h[0] = *(const v8bf*)bp;
      bfrag[nt].h[1] = *(const v8bf*)(bp + 8);
    }
#pragma unroll
    for (int mt = 0; mt < 2; ++mt) {
      V16U af;
      const __bf16* ap = Alds + ((wave * 2 + mt) * 16 + l16) * 32 + 16 * half;
      af.h[0] = *(const v8bf*)ap;
      af.h[1] = *(const v8bf*)(ap + 8);
#pragma unroll
      for (int nt = 0; nt < 2; ++nt)
        acc[mt][nt] = wmma_bf16(af.v, bfrag[nt].v, acc[mt][nt]);
    }
  }

#pragma unroll
  for (int mt = 0; mt < 2; ++mt)
#pragma unroll
    for (int nt = 0; nt < 2; ++nt)
#pragma unroll
      for (int j = 0; j < 8; ++j) {
        int row = (wave * 2 + mt) * 16 + j + 8 * half;
        int col = col0 + nt * 16 + l16;
        C[(size_t)row * N + col] = acc[mt][nt][j];
      }
}

// ---------------------------------------------------------------------------
// GPT-NeoX RoPE: pos = P + qpos; pairs (d, d+64); fp32 math, bf16 out.
// ---------------------------------------------------------------------------
__global__ void rope_kernel(const float* __restrict__ in, __bf16* __restrict__ out,
                            int heads) {
  int idx = blockIdx.x * blockDim.x + threadIdx.x;
  int total = ROWS * heads * 64;
  if (idx >= total) return;
  int d2   = idx & 63;
  int head = (idx >> 6) % heads;
  int row  = idx / (64 * heads);
  int pos  = P_ + (row & 15);
  float inv = __powf(10000.0f, -(float)d2 * (1.0f / 64.0f));
  float ang = (float)pos * inv;
  float c = cosf(ang), s = sinf(ang);
  size_t base = (size_t)row * heads * 128 + head * 128 + d2;
  float x1 = in[base], x2 = in[base + 64];
  out[base]      = (__bf16)(x1 * c - x2 * s);
  out[base + 64] = (__bf16)(x2 * c + x1 * s);
}

__global__ void cvt_kernel(const float* __restrict__ in, __bf16* __restrict__ out,
                           int n) {
  int i = blockIdx.x * blockDim.x + threadIdx.x;
  if (i < n) out[i] = (__bf16)in[i];
}

// ---------------------------------------------------------------------------
// Flash-decoding attention. grid = B*NKV*NSPLIT (256 blocks), 4 waves/block.
// TDM streams next K/V fp32 tile into the spare LDS staging buffer while the
// block converts + WMMAs the current one. Wave w = GQA group g: 16 query rows.
// ---------------------------------------------------------------------------
__global__ __launch_bounds__(128)
void flash_attn(const float* __restrict__ past_key,
                const float* __restrict__ past_value,
                const float* __restrict__ bias,
                const __bf16* __restrict__ qbf, const __bf16* __restrict__ kbf,
                const __bf16* __restrict__ vbf,
                float* __restrict__ pO, float* __restrict__ pM,
                float* __restrict__ pL) {
  __shared__ __attribute__((aligned(16))) float  Kst[2][KV_TILE * 128];  // TDM dst
  __shared__ __attribute__((aligned(16))) float  Vst[2][KV_TILE * 128];  // TDM dst
  __shared__ __attribute__((aligned(16))) __bf16 Klds[KV_TILE * 128];
  __shared__ __attribute__((aligned(16))) __bf16 VT[128 * 72];   // padded stride
  __shared__ __attribute__((aligned(16))) __bf16 Plds[4][16 * 64];

  const int blk   = blockIdx.x;
  const int split = blk & (NSPLIT - 1);
  const int bk    = blk / NSPLIT;
  const int kv    = bk % NKV_, b = bk / NKV_;
  const int tid   = threadIdx.x;
  const int wave  = tid >> 5, lane = tid & 31;
  const int l16   = lane & 15, half = lane >> 4;

  const float* kcache = past_key   + (((size_t)b * NKV_ + kv) * P_) * D_;
  const float* vcache = past_value + (((size_t)b * NKV_ + kv) * P_) * D_;

  // Q fragments (A layout, 16x32 per chunk), resident in VGPRs for the kernel.
  V16U qf[4];
  {
    const __bf16* qp =
        qbf + (size_t)(b * Q_ + l16) * (NH_ * D_) + (kv * GQA + wave) * D_;
#pragma unroll
    for (int c = 0; c < 4; ++c) {
      int b0 = 32 * c + 16 * half;
      qf[c].h[0] = *(const v8bf*)(qp + b0);
      qf[c].h[1] = *(const v8bf*)(qp + b0 + 8);
    }
  }

  v8f O[8] = {};
  float m[8], l[8];
#pragma unroll
  for (int j = 0; j < 8; ++j) { m[j] = -1e30f; l[j] = 0.f; }
  const float scale = 0.08838834764831845f;  // 1/sqrt(128)

  int t0 = split * TPS; if (t0 > NTILES) t0 = NTILES;
  int t1 = t0 + TPS;    if (t1 > NTILES) t1 = NTILES;
  const int n = t1 - t0;

  // Prologue: DMA tile t0 into staging buffer 0 (t0 is always fully in-cache).
  if (wave == 0) {
    tdm_load_tile_f32((unsigned)(size_t)(const void*)&Kst[0][0],
                      kcache + (size_t)(t0 * KV_TILE) * D_);
    tdm_load_tile_f32((unsigned)(size_t)(const void*)&Vst[0][0],
                      vcache + (size_t)(t0 * KV_TILE) * D_);
  }

  for (int i = 0; i < n; ++i) {
    const int t = t0 + i;
    const int buf = i & 1;
    const int sbase = t * KV_TILE;
    // Issue DMA for the next tile into the other buffer, then wait for the
    // current tile's two transfers (2 newest may remain outstanding).
    if (wave == 0) {
      bool nxt = false;
      if (i + 1 < n) {
        int sb2 = (t + 1) * KV_TILE;
        if (sb2 + KV_TILE <= P_) {   // fully inside the past cache
          tdm_load_tile_f32((unsigned)(size_t)(const void*)&Kst[buf ^ 1][0],
                            kcache + (size_t)sb2 * D_);
          tdm_load_tile_f32((unsigned)(size_t)(const void*)&Vst[buf ^ 1][0],
                            vcache + (size_t)sb2 * D_);
          nxt = true;
        }
      }
      if (nxt) __builtin_amdgcn_s_wait_tensorcnt(2);
      else     __builtin_amdgcn_s_wait_tensorcnt(0);
    }
    __syncthreads();   // TDM data visible; prev compute done with Klds/VT

    // Convert staging fp32 -> bf16 K (row-major) and V^T (dim-major).
    // Tail rows (fresh RoPE'd keys, s >= P) come straight from global bf16.
    for (int idx = tid; idx < KV_TILE * 128; idx += 128) {
      int key = idx >> 7, d = idx & 127;
      int s = sbase + key;
      float kval, vval;
      if (s < P_) {
        kval = Kst[buf][idx];
        vval = Vst[buf][idx];
      } else if (s < S_) {
        size_t widx = (size_t)(b * Q_ + (s - P_)) * (NKV_ * D_) + kv * D_ + d;
        kval = (float)kbf[widx];
        vval = (float)vbf[widx];
      } else {
        kval = 0.f; vval = 0.f;
      }
      Klds[idx]        = (__bf16)kval;
      VT[d * 72 + key] = (__bf16)vval;
    }
    __syncthreads();

    // scores: 16 rows x 64 keys, K-dim = D in 4 chunks of 32
    v8f sc[4];
#pragma unroll
    for (int nt = 0; nt < 4; ++nt) {
      v8f a = {};
#pragma unroll
      for (int c = 0; c < 4; ++c) {
        V16U kf;  // B layout: lane half selects K 0-15 / 16-31 of the chunk
        const __bf16* kp = Klds + (nt * 16 + l16) * 128 + c * 32 + 16 * half;
        kf.h[0] = *(const v8bf*)kp;
        kf.h[1] = *(const v8bf*)(kp + 8);
        a = wmma_bf16(qf[c].v, kf.v, a);
      }
      sc[nt] = a;
    }

    // online softmax; C-fragment row of VGPR j is qpos = j + 8*half
#pragma unroll
    for (int j = 0; j < 8; ++j) {
      int qpos = j + 8 * half;
      float mx = m[j];
#pragma unroll
      for (int nt = 0; nt < 4; ++nt) {
        int col = sbase + nt * 16 + l16;
        float x = sc[nt][j] * scale;
        x = (col < S_) ? (x + bias[((size_t)b * Q_ + qpos) * S_ + col]) : -1e30f;
        sc[nt][j] = x;
        mx = fmaxf(mx, x);
      }
      for (int off = 1; off < 16; off <<= 1) mx = fmaxf(mx, __shfl_xor(mx, off));
      float corr = __expf(m[j] - mx);
      m[j] = mx;
      float ps = 0.f;
#pragma unroll
      for (int nt = 0; nt < 4; ++nt) {
        float p = __expf(sc[nt][j] - mx);
        Plds[wave][qpos * 64 + nt * 16 + l16] = (__bf16)p;
        ps += p;
      }
      for (int off = 1; off < 16; off <<= 1) ps += __shfl_xor(ps, off);
      l[j] = l[j] * corr + ps;
#pragma unroll
      for (int ot = 0; ot < 8; ++ot) O[ot][j] *= corr;
    }

    // O += P @ V  (K-dim = 64 keys in 2 chunks of 32, N = 128 dims in 8 tiles)
#pragma unroll
    for (int kk = 0; kk < 2; ++kk) {
      V16U pf;  // A layout from the per-wave staging buffer
      const __bf16* pp = &Plds[wave][l16 * 64 + kk * 32 + 16 * half];
      pf.h[0] = *(const v8bf*)pp;
      pf.h[1] = *(const v8bf*)(pp + 8);
#pragma unroll
      for (int ot = 0; ot < 8; ++ot) {
        V16U vf;  // B layout from V^T: contiguous over keys
        const __bf16* vp = VT + (ot * 16 + l16) * 72 + kk * 32 + 16 * half;
        vf.h[0] = *(const v8bf*)vp;
        vf.h[1] = *(const v8bf*)(vp + 8);
        O[ot] = wmma_bf16(pf.v, vf.v, O[ot]);
      }
    }
  }

  // write split partials (unnormalized O, running m and l)
  size_t base = (size_t)blk * 64 * 128;
#pragma unroll
  for (int ot = 0; ot < 8; ++ot)
#pragma unroll
    for (int j = 0; j < 8; ++j) {
      int row = wave * 16 + j + 8 * half;
      pO[base + (size_t)row * 128 + ot * 16 + l16] = O[ot][j];
    }
  if (l16 == 0)
#pragma unroll
    for (int j = 0; j < 8; ++j) {
      int row = wave * 16 + j + 8 * half;
      pM[blk * 64 + row] = m[j];
      pL[blk * 64 + row] = l[j];
    }
}

// ---------------------------------------------------------------------------
// Merge the NSPLIT partials (log-sum-exp), emit bf16 attention output
// in [row = b*16+qpos][head*128+d] layout for the final projection.
// ---------------------------------------------------------------------------
__global__ __launch_bounds__(128)
void combine_kernel(const float* __restrict__ pO, const float* __restrict__ pM,
                    const float* __restrict__ pL, __bf16* __restrict__ attn) {
  int bk = blockIdx.x;
  int b = bk / NKV_, kv = bk % NKV_;
  __shared__ float coef[NSPLIT][64];
  int tid = threadIdx.x;
  if (tid < 64) {
    int row = tid;
    float ms[NSPLIT], mmax = -1e30f;
#pragma unroll
    for (int s = 0; s < NSPLIT; ++s) {
      ms[s] = pM[(bk * NSPLIT + s) * 64 + row];
      mmax = fmaxf(mmax, ms[s]);
    }
    float denom = 0.f;
#pragma unroll
    for (int s = 0; s < NSPLIT; ++s)
      denom += __expf(ms[s] - mmax) * pL[(bk * NSPLIT + s) * 64 + row];
    float inv = 1.0f / denom;
#pragma unroll
    for (int s = 0; s < NSPLIT; ++s)
      coef[s][row] = __expf(ms[s] - mmax) * inv;
  }
  __syncthreads();
  for (int i = tid; i < 64 * 128; i += 128) {
    int row = i >> 7, col = i & 127;
    float acc = 0.f;
#pragma unroll
    for (int s = 0; s < NSPLIT; ++s)
      acc += coef[s][row] * pO[((size_t)(bk * NSPLIT + s) * 64 + row) * 128 + col];
    int g = row >> 4, qpos = row & 15;
    attn[((size_t)(b * Q_ + qpos)) * (NH_ * D_) + (kv * GQA + g) * D_ + col] =
        (__bf16)acc;
  }
}

// ---------------------------------------------------------------------------
extern "C" void kernel_launch(void* const* d_in, const int* in_sizes, int n_in,
                              void* d_out, int out_size, void* d_ws, size_t ws_size,
                              hipStream_t stream) {
  (void)in_sizes; (void)n_in; (void)out_size; (void)ws_size;
  const float* hidden     = (const float*)d_in[0];
  // d_in[1] = position_ids: reference uses pos = P + qpos; computed in-kernel.
  const float* past_key   = (const float*)d_in[2];
  const float* past_value = (const float*)d_in[3];
  const float* bias       = (const float*)d_in[4];
  const float* Wq         = (const float*)d_in[5];
  const float* Wk         = (const float*)d_in[6];
  const float* Wv         = (const float*)d_in[7];
  const float* Wo         = (const float*)d_in[8];
  float* out = (float*)d_out;

  char* ws = (char*)d_ws;
  size_t off = 0;
  auto alloc = [&](size_t bytes) -> void* {
    void* p = ws + off;
    off += (bytes + 255) & ~(size_t)255;
    return p;
  };
  float*  q_f    = (float*)alloc((size_t)ROWS * NH_ * D_ * 4);
  float*  k_f    = (float*)alloc((size_t)ROWS * NKV_ * D_ * 4);
  float*  v_f    = (float*)alloc((size_t)ROWS * NKV_ * D_ * 4);
  __bf16* q_b    = (__bf16*)alloc((size_t)ROWS * NH_ * D_ * 2);
  __bf16* k_b    = (__bf16*)alloc((size_t)ROWS * NKV_ * D_ * 2);
  __bf16* v_b    = (__bf16*)alloc((size_t)ROWS * NKV_ * D_ * 2);
  __bf16* attn_b = (__bf16*)alloc((size_t)ROWS * NH_ * D_ * 2);
  float*  pO     = (float*)alloc((size_t)B_ * NKV_ * NSPLIT * 64 * 128 * 4);
  float*  pM     = (float*)alloc((size_t)B_ * NKV_ * NSPLIT * 64 * 4);
  float*  pL     = (float*)alloc((size_t)B_ * NKV_ * NSPLIT * 64 * 4);

  // QKV projections (WMMA bf16, fp32 accumulate)
  gemm128_wmma<float><<<(NH_ * D_) / 32, 128, 0, stream>>>(hidden, Wq, q_f, H_, NH_ * D_);
  gemm128_wmma<float><<<(NKV_ * D_) / 32, 128, 0, stream>>>(hidden, Wk, k_f, H_, NKV_ * D_);
  gemm128_wmma<float><<<(NKV_ * D_) / 32, 128, 0, stream>>>(hidden, Wv, v_f, H_, NKV_ * D_);
  // RoPE + bf16 conversion
  rope_kernel<<<(ROWS * NH_ * 64 + 255) / 256, 256, 0, stream>>>(q_f, q_b, NH_);
  rope_kernel<<<(ROWS * NKV_ * 64 + 255) / 256, 256, 0, stream>>>(k_f, k_b, NKV_);
  cvt_kernel<<<(ROWS * NKV_ * D_ + 255) / 256, 256, 0, stream>>>(v_f, v_b, ROWS * NKV_ * D_);
  // Flash-decoding attention: TDM-double-buffered K/V streaming
  flash_attn<<<B_ * NKV_ * NSPLIT, 128, 0, stream>>>(past_key, past_value, bias,
                                                     q_b, k_b, v_b, pO, pM, pL);
  combine_kernel<<<B_ * NKV_, 128, 0, stream>>>(pO, pM, pL, attn_b);
  // Output projection -> fp32 result
  gemm128_wmma<__bf16><<<H_ / 32, 128, 0, stream>>>(attn_b, Wo, out, NH_ * D_, H_);
}
